// Attention_15436112462557
// MI455X (gfx1250) — compile-verified, matrix operand order checked by reading, not run on hardware
//
#include <hip/hip_runtime.h>

typedef __bf16 bf16_t;
typedef __attribute__((ext_vector_type(16))) __bf16 v16bf;
typedef __attribute__((ext_vector_type(8)))  __bf16 v8bf;
typedef __attribute__((ext_vector_type(2)))  __bf16 v2bf;
typedef __attribute__((ext_vector_type(8)))  float  v8f;

#define S_LEN   2048
#define DHEAD   64
#define QTILE   64            // 4 waves x 16 rows
#define KTILE   64            // keys per main-loop iteration (4 x 16-key tiles)
#define KSTRIDE 72            // bf16 elems per sK row  (144 B, 16B multiple)
#define VSTRIDE 72            // bf16 elems per sVt row (keys 0..63 + pad)
#define PSTRIDE 72            // bf16 elems per sP row

// Fully fused row-max: mnew = max(M, rowmax16(max(x0,x1,x2,x3))).
// 4-input max + 4-stage DPP16 xor-butterfly + running-max combine, all in one
// asm block so clang cannot insert canonicalizing self-max ops in between.
__device__ __forceinline__ float rowmax16(float x0, float x1, float x2, float x3,
                                          float m) {
    float r;
    asm("v_max3_num_f32 %0, %1, %2, %3\n\t"
        "v_max_num_f32_e32 %0, %4, %0\n\t"
        "v_max_num_f32_dpp %0, %0, %0 row_xmask:1 row_mask:0xf bank_mask:0xf bound_ctrl:1\n\t"
        "v_max_num_f32_dpp %0, %0, %0 row_xmask:2 row_mask:0xf bank_mask:0xf bound_ctrl:1\n\t"
        "v_max_num_f32_dpp %0, %0, %0 row_xmask:4 row_mask:0xf bank_mask:0xf bound_ctrl:1\n\t"
        "v_max_num_f32_dpp %0, %0, %0 row_xmask:8 row_mask:0xf bank_mask:0xf bound_ctrl:1\n\t"
        "v_max_num_f32_e32 %0, %5, %0"
        : "=&v"(r)
        : "v"(x0), "v"(x1), "v"(x2), "v"(x3), "v"(m));
    return r;
}
// broadcast lane 0 of each 16-lane row to the whole row (epilogue only)
__device__ __forceinline__ float bcast16(float x) {
    int r = __builtin_amdgcn_update_dpp(0, __builtin_bit_cast(int, x),
                                        0x150 /*row_share:0*/, 0xf, 0xf, true);
    return __builtin_bit_cast(float, r);
}

// Load one 16x32 bf16 WMMA operand's per-lane 16 elements from an LDS row.
// Element e (0..7)  -> K = e     + 8*half   (VGPR0-3)
// Element e (8..15) -> K = e + 8 + 8*half   (VGPR4-7)
__device__ __forceinline__ v16bf ld_op(const bf16_t* rowseg, int half) {
    v8bf lo = *(const v8bf*)(rowseg + 8 * half);
    v8bf hi = *(const v8bf*)(rowseg + 16 + 8 * half);
    v16bf r;
#pragma unroll
    for (int i = 0; i < 8; ++i) { r[i] = lo[i]; r[i + 8] = hi[i]; }
    return r;
}

// ---- cooperative staging helpers: thread owns keys {keyp,keyp+1}, dims [db,db+16) ----
__device__ __forceinline__ void stage_load(const float* kp, const float* vp,
                                           float* kf, float* vf) {
#pragma unroll
    for (int i = 0; i < 4; ++i) {
        float4 a = ((const float4*)kp)[i];
        float4 b = ((const float4*)(kp + DHEAD))[i];
        float4 c = ((const float4*)vp)[i];
        float4 d = ((const float4*)(vp + DHEAD))[i];
        kf[4*i+0]=a.x; kf[4*i+1]=a.y; kf[4*i+2]=a.z; kf[4*i+3]=a.w;
        kf[16+4*i+0]=b.x; kf[16+4*i+1]=b.y; kf[16+4*i+2]=b.z; kf[16+4*i+3]=b.w;
        vf[4*i+0]=c.x; vf[4*i+1]=c.y; vf[4*i+2]=c.z; vf[4*i+3]=c.w;
        vf[16+4*i+0]=d.x; vf[16+4*i+1]=d.y; vf[16+4*i+2]=d.z; vf[16+4*i+3]=d.w;
    }
}
__device__ __forceinline__ void stage_store(bf16_t* sKb, bf16_t* sVtb,
                                            const float* kf, const float* vf,
                                            int keyp, int db) {
    v8bf k0a, k0b, k1a, k1b;
#pragma unroll
    for (int i = 0; i < 8; ++i) {
        k0a[i] = (bf16_t)kf[i];      k0b[i] = (bf16_t)kf[8 + i];
        k1a[i] = (bf16_t)kf[16 + i]; k1b[i] = (bf16_t)kf[24 + i];
    }
    *(v8bf*)&sKb[ keyp      * KSTRIDE + db]     = k0a;
    *(v8bf*)&sKb[ keyp      * KSTRIDE + db + 8] = k0b;
    *(v8bf*)&sKb[(keyp + 1) * KSTRIDE + db]     = k1a;
    *(v8bf*)&sKb[(keyp + 1) * KSTRIDE + db + 8] = k1b;
    // V transposed: key-pair packed into one b32 store per dim
#pragma unroll
    for (int i = 0; i < 16; ++i) {
        v2bf p;
        p[0] = (bf16_t)vf[i];        // key = keyp
        p[1] = (bf16_t)vf[16 + i];   // key = keyp+1
        *(v2bf*)&sVtb[(db + i) * VSTRIDE + keyp] = p;
    }
}

__global__ __launch_bounds__(128, 2)
void fa_wmma_bf16_kernel(const float* __restrict__ q,
                         const float* __restrict__ k,
                         const float* __restrict__ v,
                         float* __restrict__ out) {
    __shared__ alignas(16) bf16_t sK [2][KTILE * KSTRIDE];   // [buf][key][dim]
    __shared__ alignas(16) bf16_t sVt[2][DHEAD * VSTRIDE];   // [buf][dim][key]
    __shared__ alignas(16) bf16_t sP [4][16 * PSTRIDE];      // per-wave P tiles

    const int tid  = threadIdx.x;
    const int wave = tid >> 5;
    const int lane = tid & 31;
    const int half = lane >> 4;
    const int n    = lane & 15;

    const int    qblk = blockIdx.x;                 // 0..31
    const int    bh   = blockIdx.y;                 // 0..31  (B*H)
    const size_t base = (size_t)bh * S_LEN * DHEAD;

    // reference scale = 1/sqrt(ndim) = 0.5 ; fold log2(e) for exp2-domain softmax
    const float sc = 0.5f * 1.44269504088896340736f;

    // ---- persistent Q tile: two 16x32 bf16 A operands, PRE-SCALED by sc ----
    const int    qrow = qblk * QTILE + wave * 16 + n;   // A-layout row m = lane%16
    const float* qp   = q + base + (size_t)qrow * DHEAD;
    v16bf a_lo, a_hi;
    {
        const float* p0 = qp +      8 * half;
        const float* p1 = qp + 16 + 8 * half;
        const float* p2 = qp + 32 + 8 * half;
        const float* p3 = qp + 48 + 8 * half;
#pragma unroll
        for (int i = 0; i < 8; ++i) {
            a_lo[i]     = (bf16_t)(p0[i] * sc);
            a_lo[i + 8] = (bf16_t)(p1[i] * sc);
            a_hi[i]     = (bf16_t)(p2[i] * sc);
            a_hi[i + 8] = (bf16_t)(p3[i] * sc);
        }
    }

    // "ones column" B tile for the row-sum accumulator: column n==0 is 1.0
    v16bf bones;
    {
        const bf16_t val = (n == 0) ? (bf16_t)1.0f : (bf16_t)0.0f;
#pragma unroll
        for (int i = 0; i < 16; ++i) bones[i] = val;
    }

    // acc[0..3] : output tiles (dim groups) ; acc[4] : row-sum (L) tile, column 0
    float M[8];
    v8f   acc[5];
#pragma unroll
    for (int j = 0; j < 8; ++j) M[j] = -3.0e38f;
#pragma unroll
    for (int t = 0; t < 5; ++t)
#pragma unroll
        for (int j = 0; j < 8; ++j) acc[t][j] = 0.0f;

    const int keyp = (tid & 31) * 2;      // this thread's key pair within block
    const int db   = (tid >> 5) * 16;     // this thread's dim chunk

    // ---- prologue: stage key-block 0 into buffer 0 ----
    {
        float kf[32], vf[32];
        const float* kp0 = k + base + (size_t)keyp * DHEAD + db;
        const float* vp0 = v + base + (size_t)keyp * DHEAD + db;
        stage_load(kp0, vp0, kf, vf);
        stage_store(sK[0], sVt[0], kf, vf, keyp, db);
    }
    __syncthreads();

    for (int kb = 0; kb < S_LEN; kb += KTILE) {
        const int  cur       = (kb >> 6) & 1;
        const bool have_next = (kb + KTILE) < S_LEN;

        // ---- issue next block's global loads early (overlap with compute) ----
        float kf[32], vf[32];
        if (have_next) {
            const float* kp = k + base + (size_t)(kb + KTILE + keyp) * DHEAD + db;
            const float* vp = v + base + (size_t)(kb + KTILE + keyp) * DHEAD + db;
            stage_load(kp, vp, kf, vf);
            if (kb + 2 * KTILE < S_LEN)   // warm L2 for the block after next
                __builtin_prefetch(k + base + (size_t)(kb + 2 * KTILE + keyp) * DHEAD + db, 0, 1);
        }

        // ---- S = Q * K^T : four 16x16 score tiles over 64 keys ----
        const bf16_t* sKc = sK[cur];
        v8f s[4];
#pragma unroll
        for (int c = 0; c < 4; ++c) {
            const bf16_t* kr  = &sKc[(n + 16 * c) * KSTRIDE];
            v16bf blo = ld_op(kr,      half);
            v16bf bhi = ld_op(kr + 32, half);
            v8f z;
#pragma unroll
            for (int j = 0; j < 8; ++j) z[j] = 0.0f;
            z = __builtin_amdgcn_wmma_f32_16x16x32_bf16(false, a_lo, false, blo, (short)0, z, false, false);
            z = __builtin_amdgcn_wmma_f32_16x16x32_bf16(false, a_hi, false, bhi, (short)0, z, false, false);
            s[c] = z;
        }

        // ---- online softmax: fused-DPP row max, row-sum via the ones-column WMMA ----
        float e[4][8], alpha[8];
#pragma unroll
        for (int j = 0; j < 8; ++j) {
            float x0 = s[0][j], x1 = s[1][j], x2 = s[2][j], x3 = s[3][j];
            float mnew = rowmax16(x0, x1, x2, x3, M[j]);
            alpha[j] = __builtin_amdgcn_exp2f(M[j] - mnew);
            e[0][j]  = __builtin_amdgcn_exp2f(x0 - mnew);
            e[1][j]  = __builtin_amdgcn_exp2f(x1 - mnew);
            e[2][j]  = __builtin_amdgcn_exp2f(x2 - mnew);
            e[3][j]  = __builtin_amdgcn_exp2f(x3 - mnew);
            M[j]     = mnew;
        }
#pragma unroll
        for (int t = 0; t < 5; ++t)         // includes the L tile
#pragma unroll
            for (int j = 0; j < 8; ++j) acc[t][j] *= alpha[j];

        // ---- P: C-layout -> A-layout via wave-private LDS tile ----
        {
            bf16_t* pw = sP[wave];
#pragma unroll
            for (int j = 0; j < 8; ++j) {
                const int r = j + 8 * half;
#pragma unroll
                for (int c = 0; c < 4; ++c)
                    pw[r * PSTRIDE + n + 16 * c] = (bf16_t)e[c][j];
            }
        }
        // wave-private: DS ops are in-order per wave, only need the counter drained
        asm volatile("s_wait_dscnt 0" ::: "memory");

        // ---- O += P * V (and L += P * ones) : K=32 WMMAs ----
        {
            const bf16_t* prow = &sP[wave][n * PSTRIDE];
            v16bf ap0 = ld_op(prow,      half);   // keys  0..31
            v16bf ap1 = ld_op(prow + 32, half);   // keys 32..63
            const bf16_t* sVc = sVt[cur];
#pragma unroll
            for (int t = 0; t < 4; ++t) {
                const bf16_t* vr = &sVc[(t * 16 + n) * VSTRIDE];
                v16bf bv0 = ld_op(vr,      half);
                v16bf bv1 = ld_op(vr + 32, half);
                acc[t] = __builtin_amdgcn_wmma_f32_16x16x32_bf16(false, ap0, false, bv0, (short)0, acc[t], false, false);
                acc[t] = __builtin_amdgcn_wmma_f32_16x16x32_bf16(false, ap1, false, bv1, (short)0, acc[t], false, false);
            }
            // row sums of P land in column 0 of acc[4] (co-executes with next iter's VALU)
            acc[4] = __builtin_amdgcn_wmma_f32_16x16x32_bf16(false, ap0, false, bones, (short)0, acc[4], false, false);
            acc[4] = __builtin_amdgcn_wmma_f32_16x16x32_bf16(false, ap1, false, bones, (short)0, acc[4], false, false);
        }

        // ---- commit next block into the other buffer, single barrier per iteration ----
        if (have_next)
            stage_store(sK[cur ^ 1], sVt[cur ^ 1], kf, vf, keyp, db);
        __syncthreads();
    }

    // ---- epilogue: broadcast L from column 0, normalize, store fp32 ----
    float invL[8];
#pragma unroll
    for (int j = 0; j < 8; ++j)
        invL[j] = __builtin_amdgcn_rcpf(bcast16(acc[4][j]));

    float* op = out + base + (size_t)(qblk * QTILE + wave * 16) * DHEAD;
#pragma unroll
    for (int j = 0; j < 8; ++j) {
        const int r = j + 8 * half;
#pragma unroll
        for (int t = 0; t < 4; ++t)
            op[(size_t)r * DHEAD + t * 16 + n] = acc[t][j] * invL[j];
    }
}

extern "C" void kernel_launch(void* const* d_in, const int* in_sizes, int n_in,
                              void* d_out, int out_size, void* d_ws, size_t ws_size,
                              hipStream_t stream) {
    const float* q = (const float*)d_in[0];
    const float* k = (const float*)d_in[1];
    const float* v = (const float*)d_in[2];
    float* out     = (float*)d_out;

    dim3 grid(S_LEN / QTILE, 2 * 16);   // (q-blocks, B*H) = (32, 32)
    dim3 block(128);                    // 4 waves (wave32)
    fa_wmma_bf16_kernel<<<grid, block, 0, stream>>>(q, k, v, out);
}
